// LatentResidualLIF_7284264534655
// MI455X (gfx1250) — matrix operand order repaired and checked by reading, not compiled, not run on hardware
//
#include <hip/hip_runtime.h>
#include <math.h>

typedef __attribute__((ext_vector_type(16))) __bf16 v16bf;
typedef __attribute__((ext_vector_type(8)))  float  v8f;

#define BATCH  2048
#define DIM    1024
#define TSTEPS 32

#define LAT_ELEMS (BATCH * DIM)   // 2M
#define W_ELEMS   (DIM * DIM)     // 1M

// ---------------------------------------------------------------------------
// Pre-pass: split f32 -> (hi, lo) bf16 pairs, each element converted ONCE.
// Covers latent (2M elems) then W (1M elems); float4 in, 2x 8B out per array.
// ---------------------------------------------------------------------------
__global__ __launch_bounds__(256)
void split_kernel(const float* __restrict__ latent,
                  const float* __restrict__ W,
                  __bf16* __restrict__ lat_hi, __bf16* __restrict__ lat_lo,
                  __bf16* __restrict__ w_hi,   __bf16* __restrict__ w_lo)
{
  const int idx = blockIdx.x * blockDim.x + threadIdx.x;   // quad index
  const int LAT4 = LAT_ELEMS / 4;
  const float* src;
  __bf16 *dh, *dl;
  int off;
  if (idx < LAT4) { src = latent; dh = lat_hi; dl = lat_lo; off = idx * 4; }
  else            { src = W;      dh = w_hi;   dl = w_lo;   off = (idx - LAT4) * 4; }

  const float4 x4 = *(const float4*)(src + off);
  const float xs[4] = {x4.x, x4.y, x4.z, x4.w};
  __bf16 h[4], l[4];
#pragma unroll
  for (int j = 0; j < 4; ++j) {
    h[j] = (__bf16)xs[j];
    l[j] = (__bf16)(xs[j] - (float)h[j]);
  }
  *(uint2*)(dh + off) = *(uint2*)h;
  *(uint2*)(dl + off) = *(uint2*)l;
}

// ---------------------------------------------------------------------------
// GEMM: current[m, n] = sum_d latent[m, d] * W[n, d] + bias[n]
// f32 via 3x bf16 WMMA (lo*hi + hi*lo + hi*hi), operands pre-split to bf16.
// Inner loop is pure loads + v_wmma (no conversion VALU).
// Block: 256 threads = 8 waves, 2 (M) x 4 (N-groups); wave tile 16x64.
// Grid: (DIM/256, BATCH/32) = (4, 64).
// ---------------------------------------------------------------------------
__global__ __launch_bounds__(256)
void gemm_wmma_kernel(const __bf16* __restrict__ lat_hi,
                      const __bf16* __restrict__ lat_lo,
                      const __bf16* __restrict__ w_hi,
                      const __bf16* __restrict__ w_lo,
                      const float* __restrict__ bias,
                      float* __restrict__ current)
{
  const int lane  = threadIdx.x & 31;
  const int wave  = threadIdx.x >> 5;
  const int wm    = wave & 1;          // M position in block
  const int wn    = wave >> 1;         // N group in block
  const int hi16  = (lane >> 4) & 1;   // lane half selects K sub-range
  const int l15   = lane & 15;

  const int mbase = blockIdx.y * 32 + wm * 16;
  const int nbase = blockIdx.x * 256 + wn * 64;

  // A fragment (16x32 bf16): lane -> row (lane&15);
  //   lanes 0-15: K 0..15 contiguous; lanes 16-31: K 16..31 contiguous.
  const int rowA = mbase + l15;
  const int aoff = hi16 * 16;
  // B fragment (32x16 bf16): lane -> column (lane&15);
  //   lanes 0-15: K 0..15 contiguous; lanes 16-31: K 16..31 contiguous.
  const int boff = hi16 * 16;

  union Frag { v16bf v; struct { uint4 q0; uint4 q1; } u; };

  v8f acc[4] = {};   // four 16x16 f32 accumulators (N sub-tiles)

  for (int k = 0; k < DIM; k += 32) {
    Frag ah, al;
    const __bf16* pah = lat_hi + (size_t)rowA * DIM + (k + aoff);
    const __bf16* pal = lat_lo + (size_t)rowA * DIM + (k + aoff);
    __builtin_prefetch(pah + 32, 0, 0);        // next K-slice -> global_prefetch_b8
    ah.u.q0 = *(const uint4*)(pah);            // K run 0 (8 bf16 = 16B)
    ah.u.q1 = *(const uint4*)(pah + 8);        // K run 1
    al.u.q0 = *(const uint4*)(pal);
    al.u.q1 = *(const uint4*)(pal + 8);

#pragma unroll
    for (int nt = 0; nt < 4; ++nt) {
      const int col = nbase + nt * 16 + l15;
      const __bf16* pbh = w_hi + (size_t)col * DIM + (k + boff);
      const __bf16* pbl = w_lo + (size_t)col * DIM + (k + boff);
      __builtin_prefetch(pbh + 32, 0, 0);
      const v16bf bh = *(const v16bf*)(pbh);   // 16 contiguous bf16 (32B)
      const v16bf bl = *(const v16bf*)(pbl);

      // acc += alo*bhi + ahi*blo + ahi*bhi  (small terms first)
      acc[nt] = __builtin_amdgcn_wmma_f32_16x16x32_bf16(
          false, al.v, false, bh, (short)0, acc[nt], false, false);
      acc[nt] = __builtin_amdgcn_wmma_f32_16x16x32_bf16(
          false, ah.v, false, bl, (short)0, acc[nt], false, false);
      acc[nt] = __builtin_amdgcn_wmma_f32_16x16x32_bf16(
          false, ah.v, false, bh, (short)0, acc[nt], false, false);
    }
  }

  // Epilogue: bias add + store.
  // C/D layout: VGPR i -> row (i + 8*hi16), column = lane&15 (per N sub-tile).
#pragma unroll
  for (int nt = 0; nt < 4; ++nt) {
    const int col = nbase + nt * 16 + l15;
    const float bv = bias[col];
#pragma unroll
    for (int i = 0; i < 8; ++i) {
      const int row = mbase + i + hi16 * 8;
      current[(size_t)row * DIM + col] = acc[nt][i] + bv;
    }
  }
}

// ---------------------------------------------------------------------------
// Fused LIF scan (32 steps) + spike store + mean + sigmoid-scaled residual
// + ReLU. One thread per 4 neurons; membrane state in registers.
// Store-bandwidth bound: 268 MB spike tensor dominates the whole problem.
// ---------------------------------------------------------------------------
__global__ __launch_bounds__(256)
void lif_fused_kernel(const float* __restrict__ latent,
                      const float* __restrict__ gain,
                      const float* __restrict__ current,
                      float* __restrict__ out)
{
  const size_t U = (size_t)BATCH * DIM;             // updated element count
  float* updated = out;
  float* spikes  = out + U;                         // [B, T, D]

  const int idx = blockIdx.x * blockDim.x + threadIdx.x;   // [0, B*D/4)
  const int dq  = idx & (DIM / 4 - 1);
  const int row = idx >> 8;                                // DIM/4 == 256
  const int d   = dq * 4;

  const float4 c4 = *(const float4*)(current + (size_t)row * DIM + d);
  const float4 l4 = *(const float4*)(latent  + (size_t)row * DIM + d);
  const float cur[4] = {c4.x, c4.y, c4.z, c4.w};
  const float lat[4] = {l4.x, l4.y, l4.z, l4.w};
  float mem[4] = {0.f, 0.f, 0.f, 0.f};
  float cnt[4] = {0.f, 0.f, 0.f, 0.f};

  float* sp = spikes + (size_t)row * (TSTEPS * DIM) + d;
#pragma unroll
  for (int t = 0; t < TSTEPS; ++t) {
    float s[4];
#pragma unroll
    for (int j = 0; j < 4; ++j) {
      const float reset = (mem[j] > 1.0f) ? 1.0f : 0.0f;   // from previous mem
      mem[j] = 0.9f * mem[j] + cur[j] - reset;             // THRESHOLD == 1
      const float spk = (mem[j] > 1.0f) ? 1.0f : 0.0f;
      cnt[j] += spk;
      s[j] = spk;
    }
    *(float4*)(sp + (size_t)t * DIM) = *(float4*)s;        // coalesced store
  }

  const float scale = 0.4f / (1.0f + __expf(-gain[0]));
  float u[4];
#pragma unroll
  for (int j = 0; j < 4; ++j) {
    const float v = lat[j] + scale * (cnt[j] * (1.0f / TSTEPS));
    u[j] = v > 0.0f ? v : 0.0f;
  }
  *(float4*)(updated + (size_t)row * DIM + d) = *(float4*)u;

  if (idx == 0)
    out[U + (size_t)BATCH * TSTEPS * DIM] = scale;         // scalar output
}

// ---------------------------------------------------------------------------
extern "C" void kernel_launch(void* const* d_in, const int* in_sizes, int n_in,
                              void* d_out, int out_size, void* d_ws, size_t ws_size,
                              hipStream_t stream) {
  const float* latent = (const float*)d_in[0];   // [2048, 1024]
  const float* W      = (const float*)d_in[1];   // [1024, 1024]
  const float* bias   = (const float*)d_in[2];   // [1024]
  const float* gain   = (const float*)d_in[3];   // scalar

  float* out = (float*)d_out;

  // Workspace layout (~20 MB total):
  //   current : 8 MB f32, then lat_hi/lat_lo (4 MB each), w_hi/w_lo (2 MB each)
  char* ws = (char*)d_ws;
  float*  current = (float*)ws;
  __bf16* lat_hi  = (__bf16*)(ws + (size_t)LAT_ELEMS * sizeof(float));
  __bf16* lat_lo  = lat_hi + LAT_ELEMS;
  __bf16* w_hi    = lat_lo + LAT_ELEMS;
  __bf16* w_lo    = w_hi + W_ELEMS;

  // 1) one-time f32 -> bf16 hi/lo split of both operands
  const int quads = (LAT_ELEMS + W_ELEMS) / 4;   // 786432
  split_kernel<<<quads / 256, 256, 0, stream>>>(latent, W, lat_hi, lat_lo, w_hi, w_lo);

  // 2) WMMA GEMM (+bias) into `current`
  dim3 gemm_grid(DIM / 256, BATCH / 32);         // (4, 64)
  gemm_wmma_kernel<<<gemm_grid, 256, 0, stream>>>(lat_hi, lat_lo, w_hi, w_lo, bias, current);

  // 3) fused LIF + outputs
  const int nthreads = BATCH * DIM / 4;          // 524288
  lif_fused_kernel<<<nthreads / 256, 256, 0, stream>>>(latent, gain, current, out);
}
